// MultiHeadAttn_36240934044219
// MI455X (gfx1250) — compile-verified, matrix operand order checked by reading, not run on hardware
//
#include <hip/hip_runtime.h>

#define S_LEN   2048
#define D_MODEL 1024
#define BATCH   8
#define M_ROWS  (BATCH * S_LEN)  // 16384

typedef __bf16 bf16;
typedef __attribute__((ext_vector_type(16))) __bf16 v16bf;
typedef __attribute__((ext_vector_type(8)))  __bf16 bf16x8;
typedef __attribute__((ext_vector_type(8)))  float   v8f;
typedef unsigned int u32x4 __attribute__((ext_vector_type(4)));
typedef int          i32x8 __attribute__((ext_vector_type(8)));
typedef int          i32x4 __attribute__((ext_vector_type(4)));

static __device__ __forceinline__ bf16 f2bf(float f) {
  union { float f; unsigned u; } v; v.f = f;
  unsigned r = v.u + 0x7FFFu + ((v.u >> 16) & 1u);  // round-to-nearest-even
  union { unsigned short s; bf16 b; } o;
  o.s = (unsigned short)(r >> 16);
  return o.b;
}

static __device__ __forceinline__ v16bf mk_frag(bf16x8 lo, bf16x8 hip8) {
  v16bf a;
#pragma unroll
  for (int i = 0; i < 8; ++i) { a[i] = lo[i]; a[i + 8] = hip8[i]; }
  return a;
}

// Per-lane bf16 fragment per CDNA5 16-bit A/B layout:
// elements 0..7 = K (khi+0..7), elements 8..15 = K (16+khi+0..7), khi=(lane>>4)*8.
static __device__ __forceinline__ v16bf load_frag(const bf16* rowK0, int khi) {
  bf16x8 lo = *(const bf16x8*)(rowK0 + khi);
  bf16x8 hp = *(const bf16x8*)(rowK0 + 16 + khi);
  return mk_frag(lo, hp);
}

static __device__ __forceinline__ v8f wmma_bf16(v16bf a, v16bf b, v8f c) {
  return __builtin_amdgcn_wmma_f32_16x16x32_bf16(false, a, false, b, (short)0, c,
                                                 false, false);
}

// ---------------------------------------------------------------------------
// TDM: linear global->LDS copy via Tensor Data Mover (ISA ch.8 D# descriptor).
// data_size=3 (8-byte units); 1-D tile of n_units; tensor_dim0 == tile_dim0 ==
// stride so no OOB clamping. lds_off is the byte offset into this workgroup's
// LDS (dynamic LDS starts at 0 here: no static __shared__ in the kernel).
// Issued wave-wide (EXEC ignored by TDM); tracked by TENSORcnt.
// ---------------------------------------------------------------------------
static __device__ __forceinline__ void tdm_load_linear(const void* gaddr,
                                                       unsigned lds_off,
                                                       unsigned n_units) {
  unsigned long long ga = (unsigned long long)gaddr;
  u32x4 g0;
  g0.x = 1u;                                                  // count=1
  g0.y = lds_off;                                             // lds_addr
  g0.z = (unsigned)ga;                                        // global[31:0]
  g0.w = (unsigned)((ga >> 32) & 0x01FFFFFFu) | (2u << 30);   // global[56:32], type=2
  i32x8 g1;
  g1[0] = (int)(3u << 16);                       // data_size=3 (8B), mask=0
  g1[1] = (int)((n_units & 0xFFFFu) << 16);      // tensor_dim0[15:0]
  g1[2] = (int)(((n_units >> 16) & 0xFFFFu) | (1u << 16));  // dim0[31:16], dim1=1
  g1[3] = (int)((n_units & 0xFFFFu) << 16);      // tile_dim0 (<=65535)
  g1[4] = 0;                                     // tile_dim1=0 (1-D), tile_dim2=0
  g1[5] = (int)n_units;                          // tensor_dim0_stride[31:0]
  g1[6] = 0;                                     // stride0[47:32], stride1[15:0]
  g1[7] = 0;
  i32x4 z4;
  z4[0] = 0; z4[1] = 0; z4[2] = 0; z4[3] = 0;
  asm volatile("tensor_load_to_lds %0, %1, %2, %3"
               :: "s"(g0), "s"(g1), "s"(z4), "s"(z4)
               : "memory");
}

static __device__ __forceinline__ void wait_tensorcnt0() {
  asm volatile("s_wait_tensorcnt 0x0" ::: "memory");
}

// ---------------------------------------------------------------------------
// GEMM: Out[bf16] = X[fp32] @ W[fp32] + Bias[fp32]   (M=16384, N=K=1024)
// Block tile 128x128x32, 8 waves arranged 4x2, each wave 32x64 (2x4 WMMA tiles)
// ---------------------------------------------------------------------------
#define BM 128
#define BN 128
#define BK 32
#define LDT 40  // padded K-stride (40*2B = 80B, multiple of 16 -> aligned b128)

__global__ __launch_bounds__(256) void k_gemm_bias_bf16(
    const float* __restrict__ X, const float* __restrict__ W,
    const float* __restrict__ Bias, bf16* __restrict__ Out) {
  __shared__ __align__(16) bf16 As[BM * LDT];  // [m][k]
  __shared__ __align__(16) bf16 Bs[BN * LDT];  // transposed: [n][k]
  const int tid = threadIdx.x;
  const int lane = tid & 31, wid = tid >> 5;
  const int hi = (lane >> 4) * 8, l15 = lane & 15;
  const int m0 = blockIdx.y * BM, n0 = blockIdx.x * BN;
  const int wm = (wid >> 1) * 32, wn = (wid & 1) * 64;

  v8f acc[2][4];
#pragma unroll
  for (int mi = 0; mi < 2; ++mi)
#pragma unroll
    for (int ni = 0; ni < 4; ++ni)
#pragma unroll
      for (int r = 0; r < 8; ++r) acc[mi][ni][r] = 0.0f;

  for (int kb = 0; kb < D_MODEL; kb += BK) {
    if (kb + BK < D_MODEL) {  // prefetch next tiles -> global_prefetch_b8
      __builtin_prefetch(&X[(size_t)(m0 + (tid >> 1)) * D_MODEL + kb + BK +
                            (tid & 1) * 16], 0, 1);
      __builtin_prefetch(&W[(size_t)(kb + BK + (tid >> 3)) * D_MODEL + n0 +
                            (tid & 7) * 16], 0, 1);
    }
#pragma unroll
    for (int i = 0; i < 16; ++i) {  // stage A 128x32, fp32->bf16, coalesced
      int e = tid + i * 256;
      int r = e >> 5, c = e & 31;
      As[r * LDT + c] = f2bf(X[(size_t)(m0 + r) * D_MODEL + kb + c]);
    }
#pragma unroll
    for (int i = 0; i < 16; ++i) {  // stage W 32x128 transposed -> [n][k]
      int e = tid + i * 256;
      int k = e >> 7, nn = e & 127;
      Bs[nn * LDT + k] = f2bf(W[(size_t)(kb + k) * D_MODEL + n0 + nn]);
    }
    __syncthreads();
    v16bf af[2], bfr[4];
#pragma unroll
    for (int mi = 0; mi < 2; ++mi)
      af[mi] = load_frag(&As[(wm + mi * 16 + l15) * LDT], hi);
#pragma unroll
    for (int ni = 0; ni < 4; ++ni)
      bfr[ni] = load_frag(&Bs[(wn + ni * 16 + l15) * LDT], hi);
#pragma unroll
    for (int mi = 0; mi < 2; ++mi)
#pragma unroll
      for (int ni = 0; ni < 4; ++ni)
        acc[mi][ni] = wmma_bf16(af[mi], bfr[ni], acc[mi][ni]);
    __syncthreads();
  }
  // C layout: VGPR r -> row r + 8*(lane>>4), col = lane&15
#pragma unroll
  for (int mi = 0; mi < 2; ++mi)
#pragma unroll
    for (int ni = 0; ni < 4; ++ni)
#pragma unroll
      for (int r = 0; r < 8; ++r) {
        int row = m0 + wm + mi * 16 + r + hi;
        int col = n0 + wn + ni * 16 + l15;
        size_t idx = (size_t)row * D_MODEL + col;
        Out[idx] = f2bf(acc[mi][ni][r] + Bias[idx]);
      }
}

// ---------------------------------------------------------------------------
// Flash attention: one block = (batch b, 16 query rows). 8 waves, each owns a
// 128-wide slice of d. Online softmax; scores never hit global memory.
// Q tile staged by the Tensor Data Mover.
// ---------------------------------------------------------------------------
__global__ __launch_bounds__(256) void k_flash_attn(
    const bf16* __restrict__ Q, const bf16* __restrict__ K,
    const bf16* __restrict__ V, bf16* __restrict__ O,
    const int* __restrict__ nptr) {
  extern __shared__ __align__(16) char smem[];
  bf16*  Qs    = (bf16*)smem;                        // [16][1024]  (32 KB)
  bf16*  Vt    = (bf16*)(smem + 32768);              // [1024][40] d-major (80 KB)
  float* Spart = (float*)(smem + 32768 + 81920);     // [8][16][32] (16 KB)
  bf16*  Ps    = (bf16*)(smem + 32768 + 81920 + 16384);          // [16][40]
  float* crl   = (float*)(smem + 32768 + 81920 + 16384 + 1280);  // [16]

  const int n   = nptr[0];
  const int b   = blockIdx.y;
  const int q0  = blockIdx.x * 16;  // query position within sequence
  const int tid = threadIdx.x, lane = tid & 31, wid = tid >> 5;
  const int hi = (lane >> 4) * 8, l15 = lane & 15;
  const int d0 = wid * 128;  // this wave's d-slice

  const bf16* Qg = Q + ((size_t)b * S_LEN + q0) * D_MODEL;
  const bf16* Kg = K + (size_t)b * S_LEN * D_MODEL;
  const bf16* Vg = V + (size_t)b * S_LEN * D_MODEL;

  // Async-stage Q (16x1024 bf16, rows contiguous in global) via TDM:
  // 32 KB = 4096 x 8-byte units into LDS offset 0.
  if (wid == 0) tdm_load_linear(Qg, 0u, (16u * D_MODEL * 2u) / 8u);
  wait_tensorcnt0();  // waves with no outstanding tensor ops pass immediately

  v8f oacc[8];
#pragma unroll
  for (int nt = 0; nt < 8; ++nt)
#pragma unroll
    for (int r = 0; r < 8; ++r) oacc[nt][r] = 0.0f;

  float m_row = -INFINITY, l_row = 0.0f;  // per-row state, valid for tid<16

  for (int kv0 = 0; kv0 < S_LEN; kv0 += 32) {
    if (kv0 + 32 < S_LEN) {  // prefetch next K/V tiles
      __builtin_prefetch(Vg + (size_t)(kv0 + 32 + (tid >> 3)) * D_MODEL +
                         (tid & 7) * 128, 0, 1);
      __builtin_prefetch(Kg + (size_t)(kv0 + 32 + (tid >> 3)) * D_MODEL +
                         (tid & 7) * 128, 0, 1);
    }
    // stage V tile [32 kv][1024 d] transposed -> Vt[d][kv], coalesced reads
#pragma unroll 4
    for (int i = 0; i < 128; ++i) {
      int e = tid + i * 256;
      int k = e >> 10, c = e & 1023;
      Vt[c * 40 + k] = Vg[(size_t)(kv0 + k) * D_MODEL + c];
    }
    __syncthreads();  // publishes Vt (and, on iter 0, the TDM-staged Qs)

    // GEMM1: partial S[16q x 32kv] over this wave's 128-wide d-slice.
    // B = K^T -> lane col = kv row of K; d contiguous -> direct global b128s.
    v8f sc[2];
#pragma unroll
    for (int sub = 0; sub < 2; ++sub)
#pragma unroll
      for (int r = 0; r < 8; ++r) sc[sub][r] = 0.0f;
#pragma unroll
    for (int ks = 0; ks < 4; ++ks) {
      v16bf aq = load_frag(&Qs[l15 * 1024 + d0 + ks * 32], hi);
#pragma unroll
      for (int sub = 0; sub < 2; ++sub) {
        const bf16* krow =
            Kg + (size_t)(kv0 + sub * 16 + l15) * D_MODEL + d0 + ks * 32;
        v16bf bk = load_frag(krow, hi);
        sc[sub] = wmma_bf16(aq, bk, sc[sub]);
      }
    }
#pragma unroll
    for (int sub = 0; sub < 2; ++sub)
#pragma unroll
      for (int r = 0; r < 8; ++r)
        Spart[wid * 512 + (r + hi) * 32 + sub * 16 + l15] = sc[sub][r];
    __syncthreads();

    // Online softmax: 16 lanes each own one query row
    if (tid < 16) {
      const bool qv = (q0 + tid) < n;
      float mx = m_row;
      float sv[32];
#pragma unroll
      for (int j = 0; j < 32; ++j) {
        float s = 0.0f;
#pragma unroll
        for (int w = 0; w < 8; ++w) s += Spart[w * 512 + tid * 32 + j];
        s *= 0.125f;  // 1/sqrt(D_K=64)
        if (!qv || (kv0 + j) >= n) s = -1e9f;
        sv[j] = s;
        mx = fmaxf(mx, s);
      }
      float cr = __expf(m_row - mx);
      float ls = 0.0f;
#pragma unroll
      for (int j = 0; j < 32; ++j) {
        float p = __expf(sv[j] - mx);
        ls += p;
        Ps[tid * 40 + j] = f2bf(p);
      }
      m_row = mx;
      l_row = l_row * cr + ls;
      crl[tid] = cr;
    }
    __syncthreads();

    // rescale accumulators, then GEMM2: O += P[16x32] @ V[32 x 128-slice]
#pragma unroll
    for (int r = 0; r < 8; ++r) {
      float c = crl[r + hi];
#pragma unroll
      for (int nt = 0; nt < 8; ++nt) oacc[nt][r] *= c;
    }
    v16bf ap = load_frag(&Ps[l15 * 40], hi);
#pragma unroll
    for (int nt = 0; nt < 8; ++nt) {
      v16bf bv = load_frag(&Vt[(d0 + nt * 16 + l15) * 40], hi);
      oacc[nt] = wmma_bf16(ap, bv, oacc[nt]);
    }
    __syncthreads();
  }

  if (tid < 16) crl[tid] = 1.0f / l_row;
  __syncthreads();

  bf16* Og = O + ((size_t)b * S_LEN + q0) * D_MODEL;
#pragma unroll
  for (int r = 0; r < 8; ++r) {
    float invl = crl[r + hi];
#pragma unroll
    for (int nt = 0; nt < 8; ++nt) {
      int col = d0 + nt * 16 + l15;
      Og[(size_t)(r + hi) * D_MODEL + col] = f2bf(oacc[nt][r] * invl);
    }
  }
}

// ---------------------------------------------------------------------------
// Output projection: Y[fp32] = O[bf16] @ Wo + bo + x ; fused per-batch sum/sumsq
// ---------------------------------------------------------------------------
__global__ __launch_bounds__(256) void k_out_proj(
    const bf16* __restrict__ A, const float* __restrict__ W,
    const float* __restrict__ Bias, const float* __restrict__ Xres,
    float* __restrict__ Y, float* __restrict__ stats) {
  __shared__ __align__(16) bf16 As[BM * LDT];
  __shared__ __align__(16) bf16 Bs[BN * LDT];
  __shared__ float red[16];
  const int tid = threadIdx.x;
  const int lane = tid & 31, wid = tid >> 5;
  const int hi = (lane >> 4) * 8, l15 = lane & 15;
  const int m0 = blockIdx.y * BM, n0 = blockIdx.x * BN;
  const int wm = (wid >> 1) * 32, wn = (wid & 1) * 64;

  v8f acc[2][4];
#pragma unroll
  for (int mi = 0; mi < 2; ++mi)
#pragma unroll
    for (int ni = 0; ni < 4; ++ni)
#pragma unroll
      for (int r = 0; r < 8; ++r) acc[mi][ni][r] = 0.0f;

  for (int kb = 0; kb < D_MODEL; kb += BK) {
    if (kb + BK < D_MODEL) {
      __builtin_prefetch(&A[(size_t)(m0 + (tid >> 1)) * D_MODEL + kb + BK], 0, 1);
      __builtin_prefetch(&W[(size_t)(kb + BK + (tid >> 3)) * D_MODEL + n0 +
                            (tid & 7) * 16], 0, 1);
    }
#pragma unroll
    for (int i = 0; i < 16; ++i) {
      int e = tid + i * 256;
      int r = e >> 5, c = e & 31;
      As[r * LDT + c] = A[(size_t)(m0 + r) * D_MODEL + kb + c];  // already bf16
    }
#pragma unroll
    for (int i = 0; i < 16; ++i) {
      int e = tid + i * 256;
      int k = e >> 7, nn = e & 127;
      Bs[nn * LDT + k] = f2bf(W[(size_t)(kb + k) * D_MODEL + n0 + nn]);
    }
    __syncthreads();
    v16bf af[2], bfr[4];
#pragma unroll
    for (int mi = 0; mi < 2; ++mi)
      af[mi] = load_frag(&As[(wm + mi * 16 + l15) * LDT], hi);
#pragma unroll
    for (int ni = 0; ni < 4; ++ni)
      bfr[ni] = load_frag(&Bs[(wn + ni * 16 + l15) * LDT], hi);
#pragma unroll
    for (int mi = 0; mi < 2; ++mi)
#pragma unroll
      for (int ni = 0; ni < 4; ++ni)
        acc[mi][ni] = wmma_bf16(af[mi], bfr[ni], acc[mi][ni]);
    __syncthreads();
  }

  float lsum = 0.0f, lsq = 0.0f;
#pragma unroll
  for (int mi = 0; mi < 2; ++mi)
#pragma unroll
    for (int ni = 0; ni < 4; ++ni)
#pragma unroll
      for (int r = 0; r < 8; ++r) {
        int row = m0 + wm + mi * 16 + r + hi;
        int col = n0 + wn + ni * 16 + l15;
        size_t idx = (size_t)row * D_MODEL + col;
        float v = acc[mi][ni][r] + Bias[idx] + Xres[idx];
        Y[idx] = v;
        lsum += v;
        lsq += v * v;
      }
  // wave reduce (wave32), then block reduce, then global atomics
#pragma unroll
  for (int off = 16; off > 0; off >>= 1) {
    lsum += __shfl_xor(lsum, off, 32);
    lsq  += __shfl_xor(lsq, off, 32);
  }
  if (lane == 0) { red[wid * 2] = lsum; red[wid * 2 + 1] = lsq; }
  __syncthreads();
  if (tid == 0) {
    float ts = 0.0f, tq = 0.0f;
#pragma unroll
    for (int w = 0; w < 8; ++w) { ts += red[w * 2]; tq += red[w * 2 + 1]; }
    const int batch = m0 >> 11;  // 2048 rows per batch, BM=128 divides evenly
    atomicAdd(&stats[batch * 2], ts);
    atomicAdd(&stats[batch * 2 + 1], tq);
  }
}

__global__ void k_init_stats(float* stats) {
  if (threadIdx.x < 16) stats[threadIdx.x] = 0.0f;
}

// LayerNorm over the whole [S,D] slab per batch, in place on Y
__global__ __launch_bounds__(256) void k_layernorm(
    float* __restrict__ Y, const float* __restrict__ gamma,
    const float* __restrict__ beta, const float* __restrict__ stats) {
  const size_t SD = (size_t)S_LEN * D_MODEL;
  size_t gi = ((size_t)blockIdx.x * 256 + threadIdx.x) * 4;
  int b = (int)(gi / SD);
  size_t sd = gi % SD;
  float invN = 1.0f / (float)SD;
  float mean = stats[b * 2] * invN;
  float var = stats[b * 2 + 1] * invN - mean * mean;
  float rstd = rsqrtf(var + 1e-5f);
  float4 y = *(float4*)(Y + gi);
  float4 g = *(const float4*)(gamma + sd);
  float4 be = *(const float4*)(beta + sd);
  y.x = (y.x - mean) * rstd * g.x + be.x;
  y.y = (y.y - mean) * rstd * g.y + be.y;
  y.z = (y.z - mean) * rstd * g.z + be.z;
  y.w = (y.w - mean) * rstd * g.w + be.w;
  *(float4*)(Y + gi) = y;
}

extern "C" void kernel_launch(void* const* d_in, const int* in_sizes, int n_in,
                              void* d_out, int out_size, void* d_ws,
                              size_t ws_size, hipStream_t stream) {
  (void)in_sizes; (void)n_in; (void)out_size; (void)ws_size;
  const float* x     = (const float*)d_in[0];
  const float* Wq    = (const float*)d_in[1];
  const float* bq    = (const float*)d_in[2];
  const float* Wk    = (const float*)d_in[3];
  const float* bk    = (const float*)d_in[4];
  const float* Wv    = (const float*)d_in[5];
  const float* bv    = (const float*)d_in[6];
  const float* Wo    = (const float*)d_in[7];
  const float* bo    = (const float*)d_in[8];
  const float* gamma = (const float*)d_in[9];
  const float* beta  = (const float*)d_in[10];
  const int*   nptr  = (const int*)d_in[11];
  float* Y = (float*)d_out;

  // workspace: Q,K,V,O (bf16, 32MB each) + stats (16 f32) = 128MB + 64B
  char* w = (char*)d_ws;
  const size_t tsz = (size_t)M_ROWS * D_MODEL * sizeof(bf16);
  bf16* Qb = (bf16*)w;  w += tsz;
  bf16* Kb = (bf16*)w;  w += tsz;
  bf16* Vb = (bf16*)w;  w += tsz;
  bf16* Ob = (bf16*)w;  w += tsz;
  float* stats = (float*)w;

  dim3 gGemm(D_MODEL / BN, M_ROWS / BM);  // (8, 128)
  k_init_stats<<<1, 32, 0, stream>>>(stats);
  k_gemm_bias_bf16<<<gGemm, 256, 0, stream>>>(x, Wq, bq, Qb);
  k_gemm_bias_bf16<<<gGemm, 256, 0, stream>>>(x, Wk, bk, Kb);
  k_gemm_bias_bf16<<<gGemm, 256, 0, stream>>>(x, Wv, bv, Vb);

  size_t shmem = 32768          // Qs [16][1024] bf16
               + 81920          // Vt [1024][40] bf16
               + 16384          // Spart [8][16][32] f32
               + 1280           // Ps [16][40] bf16
               + 64;            // crl [16] f32
  k_flash_attn<<<dim3(S_LEN / 16, BATCH), 256, shmem, stream>>>(Qb, Kb, Vb, Ob,
                                                                nptr);
  k_out_proj<<<gGemm, 256, 0, stream>>>(Ob, Wo, bo, x, Y, stats);
  k_layernorm<<<(unsigned)((size_t)BATCH * S_LEN * D_MODEL / 4 / 256), 256, 0,
               stream>>>(Y, gamma, beta, stats);
}
